// QThreeLayer_50362786512957
// MI455X (gfx1250) — compile-verified, matrix operand order checked by reading, not run on hardware
//
#include <hip/hip_runtime.h>
#include <math.h>

typedef __attribute__((ext_vector_type(16))) _Float16 v16h;
typedef __attribute__((ext_vector_type(8)))  _Float16 v8h;
typedef __attribute__((ext_vector_type(8)))  float    v8f;

#define QMAX 31.0f
#define QMIN -32.0f

__device__ __forceinline__ float rcpf(float x) {
  return __builtin_amdgcn_rcpf(x);  // v_rcp_f32, 1 ulp
}
__device__ __forceinline__ float clipq(float v) {
  return fminf(fmaxf(v, QMIN), QMAX);
}
__device__ __forceinline__ float scale_of(unsigned bits) {
  return fmaxf(__uint_as_float(bits), 1e-8f) * (1.0f / 31.0f);
}
__device__ __forceinline__ v8f wmma16(v16h a, v16h b, v8f c) {
  // D = A(16x32 f16) x B(32x16 f16) + C(16x16 f32); exact for small-int inputs
  return __builtin_amdgcn_wmma_f32_16x16x32_f16(false, a, false, b,
                                                (short)0, c, false, false);
}
// Gather one A-operand (16x32 tile, this wave's ktile) from LDS act buffer.
// A layout: lane (h,li) holds row M=li, k = kt*32 + 2*(e&8) + 8*h + (e&7)
// -> two contiguous 16B chunks per lane.
__device__ __forceinline__ v16h gatherA(const _Float16* A, int li, int h,
                                        int cin, int kt) {
  const _Float16* p = A + li * cin + kt * 32 + 8 * h;
  v8h lo = *(const v8h*)p;
  v8h hi = *(const v8h*)(p + 16);
  return __builtin_shufflevector(lo, hi, 0, 1, 2, 3, 4, 5, 6, 7,
                                 8, 9, 10, 11, 12, 13, 14, 15);
}

// ---------------------------------------------------------------------------
// Workspace layout (d_ws):
//   unsigned g[4]   : maxabs bits for x, y1, y2, y3       (bytes   0..15)
//   float    wsf[4..7]: weight scales per layer           (bytes  16..31)
//   f16 weight tiles at byte 64: B-layout tiles of 512 halves (1KB) each
//     L1: tiles 0..3 (kt=1,nt=4)  L2: 4..7 (kt=2,nt=2)
//     L3: 8..9 (kt=1,nt=2)        L4: 10   (kt=1,nt=1)
// ---------------------------------------------------------------------------

__global__ void prep_kernel(const float* __restrict__ w1,
                            const float* __restrict__ w2,
                            const float* __restrict__ w3,
                            const float* __restrict__ w4, void* ws) {
  const int tid = threadIdx.x;
  unsigned* g = (unsigned*)ws;
  float* wsf = (float*)ws;
  _Float16* wt = (_Float16*)((char*)ws + 64);
  if (tid < 4) g[tid] = 0u;  // re-init running maxes every call

  const float* W[4] = {w1, w2, w3, w4};
  const int CIN[4] = {16, 64, 32, 32};
  const int COUT[4] = {64, 32, 32, 5};
  const int KT[4] = {1, 2, 1, 1};
  const int NT[4] = {4, 2, 2, 1};
  const int TOFF[4] = {0, 4, 8, 10};

  __shared__ float red[256];
  for (int l = 0; l < 4; ++l) {
    // per-tensor weight scale
    float v = 0.0f;
    int sz = CIN[l] * COUT[l];
    for (int i = tid; i < sz; i += 256) v = fmaxf(v, fabsf(W[l][i]));
    red[tid] = v;
    __syncthreads();
    for (int s = 128; s > 0; s >>= 1) {
      if (tid < s) red[tid] = fmaxf(red[tid], red[tid + s]);
      __syncthreads();
    }
    float wsl = fmaxf(red[0], 1e-8f) * (1.0f / 31.0f);
    float iwsl = rcpf(wsl);
    if (tid == 0) wsf[4 + l] = wsl;
    __syncthreads();
    // quantize weights into WMMA B-layout tiles (k x n, lane = column)
    int total = KT[l] * NT[l] * 512;
    for (int idx = tid; idx < total; idx += 256) {
      int e = idx & 15;
      int lane = (idx >> 4) & 31;
      int t = idx >> 9;
      int kt = t / NT[l], nt = t % NT[l];
      int h = lane >> 4, li = lane & 15;
      int k = kt * 32 + ((e & 8) << 1) + (h << 3) + (e & 7);
      int n = nt * 16 + li;
      float q = 0.0f;
      if (k < CIN[l] && n < COUT[l])
        q = clipq(rintf(W[l][n * CIN[l] + k] * iwsl));
      wt[(TOFF[l] + t) * 512 + lane * 16 + e] = (_Float16)q;
    }
    __syncthreads();
  }
}

__global__ void maxabs_kernel(const float* __restrict__ x, int n,
                              unsigned* __restrict__ g) {
  float m = 0.0f;
  for (int i = blockIdx.x * blockDim.x + threadIdx.x; i < n;
       i += gridDim.x * blockDim.x)
    m = fmaxf(m, fabsf(x[i]));
  __shared__ unsigned bm;
  if (threadIdx.x == 0) bm = 0u;
  __syncthreads();
  atomicMax(&bm, __float_as_uint(m));
  __syncthreads();
  if (threadIdx.x == 0) atomicMax(&g[0], bm);
}

// Recompute layers 1..STOP from x.  STOP<4: reduce max of layer-STOP output.
// STOP==4: write softmax(out).  256 threads = 8 waves, 16 rows per wave.
template <int STOP>
__global__ void layers_kernel(const float* __restrict__ xg,
                              const float* __restrict__ bias1,
                              const float* __restrict__ bias2,
                              const float* __restrict__ bias3,
                              const float* __restrict__ bias4, void* ws,
                              float* __restrict__ out, int B) {
  const int tid = threadIdx.x;
  const int wave = tid >> 5, lane = tid & 31;
  const int h = lane >> 4, li = lane & 15;

  unsigned* g = (unsigned*)ws;
  const float* wsf = (const float*)ws;
  const _Float16* WT = (const _Float16*)((const char*)ws + 64);
  const _Float16* W1 = WT;
  const _Float16* W2 = WT + 4 * 512;
  const _Float16* W3 = WT + 8 * 512;
  const _Float16* W4 = WT + 10 * 512;

  __shared__ unsigned bmax;
  __shared__ __align__(32) _Float16 acts[8][1024];  // 2KB per wave
  _Float16* A = acts[wave];
  if (tid == 0) bmax = 0u;
  __syncthreads();

  const int rowBase = (blockIdx.x * 8 + wave) * 16;

  // ---------------- layer 1: x(16) -> 64 ----------------
  float s0 = scale_of(g[0]);
  float is0 = rcpf(s0);
  float bs1 = wsf[4] * s0;
  float ibs1 = rcpf(bs1);
  v16h a1 = {};
  {
    int row = rowBase + li;
    if (row >= B) row = B - 1;
    const float* xr = xg + (size_t)row * 16 + h * 8;
    float4 f0 = *(const float4*)xr;
    float4 f1 = *(const float4*)(xr + 4);
    float xv[8] = {f0.x, f0.y, f0.z, f0.w, f1.x, f1.y, f1.z, f1.w};
#pragma unroll
    for (int e = 0; e < 8; ++e) a1[e] = (_Float16)clipq(rintf(xv[e] * is0));
  }
  // k1 maps raw relu'd accumulator directly to next layer's int domain:
  //   q = rint( relu(acc+bi) * b_scale / s_next ) = rint( relu(acc+bi) * k1 )
  float s1 = (STOP >= 2) ? scale_of(g[1]) : 1.0f;
  float k1 = bs1 * rcpf(s1);
  float lmax = 0.0f;
#pragma unroll
  for (int nt = 0; nt < 4; ++nt) {
    v16h b = *(const v16h*)(W1 + nt * 512 + lane * 16);
    v8f c = {};
    c = wmma16(a1, b, c);
    float bi = clipq(rintf(bias1[nt * 16 + li] * ibs1));
#pragma unroll
    for (int r = 0; r < 8; ++r) {
      float t = c[r] + bi;  // integer-valued accumulator + int bias
      if (STOP == 1)
        lmax = fmaxf(lmax, t);  // uniform *bs1 applied once at reduce
      else
        A[(r + 8 * h) * 64 + nt * 16 + li] =
            (_Float16)fminf(rintf(fmaxf(t, 0.0f) * k1), QMAX);
    }
  }
  if (STOP == 1) {
    atomicMax(&bmax, __float_as_uint(fmaxf(lmax, 0.0f) * bs1));
    __syncthreads();
    if (tid == 0) atomicMax(&g[1], bmax);
    return;
  }
  __syncthreads();

  // ---------------- layer 2: 64 -> 32 ----------------
  v16h a20 = gatherA(A, li, h, 64, 0);
  v16h a21 = gatherA(A, li, h, 64, 1);
  float bs2 = wsf[5] * s1;
  float ibs2 = rcpf(bs2);
  float s2 = (STOP >= 3) ? scale_of(g[2]) : 1.0f;
  float k2 = bs2 * rcpf(s2);
  lmax = 0.0f;
#pragma unroll
  for (int nt = 0; nt < 2; ++nt) {
    v8f c = {};
    v16h b0 = *(const v16h*)(W2 + (0 * 2 + nt) * 512 + lane * 16);
    c = wmma16(a20, b0, c);
    v16h b1 = *(const v16h*)(W2 + (1 * 2 + nt) * 512 + lane * 16);
    c = wmma16(a21, b1, c);
    float bi = clipq(rintf(bias2[nt * 16 + li] * ibs2));
#pragma unroll
    for (int r = 0; r < 8; ++r) {
      float t = c[r] + bi;
      if (STOP == 2)
        lmax = fmaxf(lmax, t);
      else
        A[(r + 8 * h) * 32 + nt * 16 + li] =
            (_Float16)fminf(rintf(fmaxf(t, 0.0f) * k2), QMAX);
    }
  }
  if (STOP == 2) {
    atomicMax(&bmax, __float_as_uint(fmaxf(lmax, 0.0f) * bs2));
    __syncthreads();
    if (tid == 0) atomicMax(&g[2], bmax);
    return;
  }
  __syncthreads();

  // ---------------- layer 3: 32 -> 32 ----------------
  v16h a3 = gatherA(A, li, h, 32, 0);
  float bs3 = wsf[6] * s2;
  float ibs3 = rcpf(bs3);
  float s3 = (STOP >= 4) ? scale_of(g[3]) : 1.0f;
  float k3 = bs3 * rcpf(s3);
  lmax = 0.0f;
#pragma unroll
  for (int nt = 0; nt < 2; ++nt) {
    v16h b = *(const v16h*)(W3 + nt * 512 + lane * 16);
    v8f c = {};
    c = wmma16(a3, b, c);
    float bi = clipq(rintf(bias3[nt * 16 + li] * ibs3));
#pragma unroll
    for (int r = 0; r < 8; ++r) {
      float t = c[r] + bi;
      if (STOP == 3)
        lmax = fmaxf(lmax, t);
      else
        A[(r + 8 * h) * 32 + nt * 16 + li] =
            (_Float16)fminf(rintf(fmaxf(t, 0.0f) * k3), QMAX);
    }
  }
  if (STOP == 3) {
    atomicMax(&bmax, __float_as_uint(fmaxf(lmax, 0.0f) * bs3));
    __syncthreads();
    if (tid == 0) atomicMax(&g[3], bmax);
    return;
  }
  __syncthreads();

  // ---------------- layer 4: 32 -> 5, then softmax ----------------
  v16h a4 = gatherA(A, li, h, 32, 0);
  float bs4 = wsf[7] * s3;
  float ibs4 = rcpf(bs4);
  v16h b4v = *(const v16h*)(W4 + lane * 16);
  v8f c4 = {};
  c4 = wmma16(a4, b4v, c4);
  float* Z = (float*)A;  // reuse this wave's LDS region: 16 rows x 5 f32
  if (li < 5) {
    float bi = clipq(rintf(bias4[li] * ibs4));
#pragma unroll
    for (int r = 0; r < 8; ++r) Z[(r + 8 * h) * 5 + li] = (c4[r] + bi) * bs4;
  }
  __syncthreads();
  if (h == 0) {
    int row = rowBase + li;
    if (row < B) {
      float z[5];
#pragma unroll
      for (int j = 0; j < 5; ++j) z[j] = Z[li * 5 + j];
      float m = z[0];
#pragma unroll
      for (int j = 1; j < 5; ++j) m = fmaxf(m, z[j]);
      float e[5], ssum = 0.0f;
#pragma unroll
      for (int j = 0; j < 5; ++j) {
        e[j] = expf(z[j] - m);
        ssum += e[j];
      }
      float inv = rcpf(ssum);
#pragma unroll
      for (int j = 0; j < 5; ++j) out[(size_t)row * 5 + j] = e[j] * inv;
    }
  }
}

extern "C" void kernel_launch(void* const* d_in, const int* in_sizes, int n_in,
                              void* d_out, int out_size, void* d_ws,
                              size_t ws_size, hipStream_t stream) {
  const float* x = (const float*)d_in[0];
  const float* w1 = (const float*)d_in[1];
  const float* b1 = (const float*)d_in[2];
  const float* w2 = (const float*)d_in[3];
  const float* b2 = (const float*)d_in[4];
  const float* w3 = (const float*)d_in[5];
  const float* b3 = (const float*)d_in[6];
  const float* w4 = (const float*)d_in[7];
  const float* b4 = (const float*)d_in[8];
  float* out = (float*)d_out;
  int B = in_sizes[0] / 16;
  int mblocks = (B + 127) / 128;

  prep_kernel<<<1, 256, 0, stream>>>(w1, w2, w3, w4, d_ws);
  maxabs_kernel<<<512, 256, 0, stream>>>(x, in_sizes[0], (unsigned*)d_ws);
  layers_kernel<1><<<mblocks, 256, 0, stream>>>(x, b1, b2, b3, b4, d_ws, out, B);
  layers_kernel<2><<<mblocks, 256, 0, stream>>>(x, b1, b2, b3, b4, d_ws, out, B);
  layers_kernel<3><<<mblocks, 256, 0, stream>>>(x, b1, b2, b3, b4, d_ws, out, B);
  layers_kernel<4><<<mblocks, 256, 0, stream>>>(x, b1, b2, b3, b4, d_ws, out, B);
}
